// GSConverterNeRFMarchingCubes_28054726377961
// MI455X (gfx1250) — matrix operand order repaired and checked by reading, not compiled, not run on hardware
//
#include <hip/hip_runtime.h>

// ---------------------------------------------------------------------------
// Instant-NGP NeRF render for MI455X (gfx1250, wave32).
//   Kernel 1: per-sample hash encode (both tables share corner hashing; the
//             float2 gathers are L2-resident: tables = 100MB < 192MB L2),
//             feats streamed to LDS, both MLPs on V_WMMA_F32_16X16X32_F16
//             (fp16 in, fp32 accumulate). Weight fragments are built inside
//             each phase scope and the hash loop is only 2x unrolled to keep
//             VGPR pressure low -> more waves/SIMD to hide gather latency.
//   Kernel 2: per-ray exclusive transmittance scan + composite.
// ---------------------------------------------------------------------------

typedef __attribute__((ext_vector_type(16))) _Float16 v16h;
typedef __attribute__((ext_vector_type(8)))  _Float16 v8h;
typedef __attribute__((ext_vector_type(2)))  _Float16 v2h;
typedef __attribute__((ext_vector_type(8)))  float    v8f;

#define LNUM   12
#define TBL    (1u << 19)
#define MSAMP  524288
#define NRAYS_ 65536

// B-fragment (32x16 fp16) for layer-1 weights w1[24x32], K padded to 32.
// ISA B layout: lane holds column n = lane&15; K = 8*(lane>>4) + e (e<8), 16 + 8*(lane>>4) + (e-8) (e>=8).
__device__ __forceinline__ v16h build_b_w1(const float* __restrict__ w1, int nb, int lane) {
    const int n  = (lane & 15) + nb * 16;
    const int kh = lane >> 4;
    v16h b = {};
#pragma unroll
    for (int e = 0; e < 16; ++e) {
        const int k = (e < 8) ? (kh * 8 + e) : (16 + kh * 8 + (e - 8));
        const float v = (k < 24) ? w1[k * 32 + n] : 0.0f;
        b[e] = (_Float16)v;
    }
    return b;
}

// B-fragment for layer-2 weights w2[32 x nout], N padded to 16.
__device__ __forceinline__ v16h build_b_w2(const float* __restrict__ w2, int nout, int lane) {
    const int n  = lane & 15;
    const int kh = lane >> 4;
    v16h b = {};
#pragma unroll
    for (int e = 0; e < 16; ++e) {
        const int k = (e < 8) ? (kh * 8 + e) : (16 + kh * 8 + (e - 8));
        const float v = (n < nout) ? w2[k * nout + n] : 0.0f;
        b[e] = (_Float16)v;
    }
    return b;
}

// A-fragment (16x32 fp16) for tile rows [tb, tb+16) from row-major LDS staging
// [sample][32 halfs]. ISA A layout => two contiguous 16B ds_load_b128 per lane.
__device__ __forceinline__ v16h load_a_frag(const _Float16* sBuf, int tb, int lane) {
    const int r = lane & 15;
    const int h = lane >> 4;
    const v8h lo = *(const v8h*)&sBuf[(tb + r) * 32 + h * 8];
    const v8h hi = *(const v8h*)&sBuf[(tb + r) * 32 + 16 + h * 8];
    v16h a = {};
#pragma unroll
    for (int e = 0; e < 8; ++e) { a[e] = lo[e]; a[8 + e] = hi[e]; }
    return a;
}

// One MLP: sFeat[256x32 fp16] --(w1, relu)--> sHid[256x32 fp16] --(w2)--> sOut.
// Each wave owns 2 tiles of 16 samples. Caller syncs before/after.
__device__ __forceinline__ void mlp_phase(const _Float16* sFeat, _Float16* sHid, float* sOut,
                                          v16h b10, v16h b11, v16h b2, int nout,
                                          int lane, int wv) {
    const int h = lane >> 4;
    const int n = lane & 15;
#pragma unroll
    for (int t = 0; t < 2; ++t) {
        const int tb = wv * 32 + t * 16;
        v16h a = load_a_frag(sFeat, tb, lane);
        v8f c0 = {}; v8f c1 = {};
        c0 = __builtin_amdgcn_wmma_f32_16x16x32_f16(false, a, false, b10, (short)0, c0, false, false);
        c1 = __builtin_amdgcn_wmma_f32_16x16x32_f16(false, a, false, b11, (short)0, c1, false, false);
#pragma unroll
        for (int rr = 0; rr < 8; ++rr) {           // D layout: M = rr + 8*h, N = n (+16)
            const int m = rr + 8 * h;
            sHid[(tb + m) * 32 + n]      = (_Float16)fmaxf(c0[rr], 0.0f);
            sHid[(tb + m) * 32 + 16 + n] = (_Float16)fmaxf(c1[rr], 0.0f);
        }
    }
    __syncthreads();
#pragma unroll
    for (int t = 0; t < 2; ++t) {
        const int tb = wv * 32 + t * 16;
        v16h a = load_a_frag(sHid, tb, lane);
        v8f c = {};
        c = __builtin_amdgcn_wmma_f32_16x16x32_f16(false, a, false, b2, (short)0, c, false, false);
        if (n < nout) {
#pragma unroll
            for (int rr = 0; rr < 8; ++rr) {
                const int m = rr + 8 * h;
                sOut[(tb + m) * nout + n] = c[rr];
            }
        }
    }
}

__global__ __launch_bounds__(256)
void nerf_sample_kernel(const float* __restrict__ rays_o,
                        const float* __restrict__ rays_d,
                        const float* __restrict__ t_starts,
                        const float* __restrict__ t_ends,
                        const int*   __restrict__ ray_indices,
                        const float* __restrict__ table_density,
                        const float* __restrict__ table_color,
                        const float* __restrict__ w1_density,
                        const float* __restrict__ w2_density,
                        const float* __restrict__ w1_color,
                        const float* __restrict__ w2_color,
                        float4* __restrict__ ws) {
    __shared__ _Float16 sFeatD[256 * 32];
    __shared__ _Float16 sFeatC[256 * 32];
    __shared__ _Float16 sHid[256 * 32];
    __shared__ float    sOut[256 * 3];

    const int tid  = threadIdx.x;
    const int lane = tid & 31;
    const int wv   = tid >> 5;
    const int gid  = blockIdx.x * 256 + tid;

    // ---- sample position ----
    const int   rix  = ray_indices[gid];
    const float ts   = t_starts[gid];
    const float te   = t_ends[gid];
    const float tmid = 0.5f * (ts + te);
    const float dt   = te - ts;
    const float px = rays_o[rix * 3 + 0] + rays_d[rix * 3 + 0] * tmid;
    const float py = rays_o[rix * 3 + 1] + rays_d[rix * 3 + 1] * tmid;
    const float pz = rays_o[rix * 3 + 2] + rays_d[rix * 3 + 2] * tmid;
    const float x0 = fminf(fmaxf((px + 1.0f) * 0.5f, 0.0f), 1.0f - 1e-6f);
    const float y0 = fminf(fmaxf((py + 1.0f) * 0.5f, 0.0f), 1.0f - 1e-6f);
    const float z0 = fminf(fmaxf((pz + 1.0f) * 0.5f, 0.0f), 1.0f - 1e-6f);

    // K-padding (feats 24..31 = 0), once.
#pragma unroll
    for (int l = LNUM; l < 16; ++l) {
        *(v2h*)&sFeatD[tid * 32 + 2 * l] = (v2h){(_Float16)0.f, (_Float16)0.f};
        *(v2h*)&sFeatC[tid * 32 + 2 * l] = (v2h){(_Float16)0.f, (_Float16)0.f};
    }

    // ---- hash-grid encode, streamed to LDS (keeps VGPR count low) ----
    const float2* tD = (const float2*)table_density;
    const float2* tC = (const float2*)table_color;
#pragma unroll 2
    for (int l = 0; l < LNUM; ++l) {
        const float res = (float)(16u << l);
        const float sx = x0 * res, sy = y0 * res, sz = z0 * res;
        const float fx0 = floorf(sx), fy0 = floorf(sy), fz0 = floorf(sz);
        const float fx = sx - fx0, fy = sy - fy0, fz = sz - fz0;
        const unsigned ux = (unsigned)fx0, uy = (unsigned)fy0, uz = (unsigned)fz0;
        const size_t lbase = (size_t)l * TBL;
        float aD0 = 0.f, aD1 = 0.f, aC0 = 0.f, aC1 = 0.f;
#pragma unroll
        for (int c = 0; c < 8; ++c) {
            const unsigned bx = c & 1u, by = (c >> 1) & 1u, bz = (c >> 2) & 1u;
            const unsigned hsh = (ux + bx) * 1u
                               ^ (uy + by) * 2654435761u
                               ^ (uz + bz) * 805459861u;
            const size_t idx = lbase + (hsh & (TBL - 1u));
            const float w = (bx ? fx : 1.0f - fx) * (by ? fy : 1.0f - fy) * (bz ? fz : 1.0f - fz);
            const float2 vD = tD[idx];
            const float2 vC = tC[idx];
            aD0 += w * vD.x; aD1 += w * vD.y;
            aC0 += w * vC.x; aC1 += w * vC.y;
        }
        *(v2h*)&sFeatD[tid * 32 + 2 * l] = (v2h){(_Float16)aD0, (_Float16)aD1};
        *(v2h*)&sFeatC[tid * 32 + 2 * l] = (v2h){(_Float16)aC0, (_Float16)aC1};
    }
    __syncthreads();

    // ===== density MLP -> sigma ===== (weight frags live only in this scope)
    {
        const v16h b0 = build_b_w1(w1_density, 0, lane);
        const v16h b1 = build_b_w1(w1_density, 1, lane);
        const v16h b2 = build_b_w2(w2_density, 1, lane);
        mlp_phase(sFeatD, sHid, sOut, b0, b1, b2, 1, lane, wv);
    }
    __syncthreads();
    const float sval = __expf(sOut[tid]) * dt;     // trunc_exp fwd == exp; s = sigma*dt

    // ===== color MLP -> rgb =====
    // (sOut reads above complete before any thread passes mlp_phase's internal
    //  barrier, which precedes the layer-2 sOut writes.)
    {
        const v16h b0 = build_b_w1(w1_color, 0, lane);
        const v16h b1 = build_b_w1(w1_color, 1, lane);
        const v16h b2 = build_b_w2(w2_color, 3, lane);
        mlp_phase(sFeatC, sHid, sOut, b0, b1, b2, 3, lane, wv);
    }
    __syncthreads();
    const float cr = 1.0f / (1.0f + __expf(-sOut[tid * 3 + 0]));
    const float cg = 1.0f / (1.0f + __expf(-sOut[tid * 3 + 1]));
    const float cb = 1.0f / (1.0f + __expf(-sOut[tid * 3 + 2]));

    ws[gid] = make_float4(sval, cr, cg, cb);
}

// Per-ray exclusive transmittance scan + composite. ray_indices is sorted,
// avg 8 samples/ray; two 19-step binary searches find [start, end).
__global__ __launch_bounds__(256)
void nerf_render_kernel(const int* __restrict__ ray_indices,
                        const float4* __restrict__ ws,
                        float* __restrict__ out) {
    const int rid = blockIdx.x * 256 + threadIdx.x;

    int lo = 0, hi = MSAMP;
    while (lo < hi) { const int mid = (lo + hi) >> 1; if (ray_indices[mid] < rid) lo = mid + 1; else hi = mid; }
    const int start = lo;
    hi = MSAMP;
    while (lo < hi) { const int mid = (lo + hi) >> 1; if (ray_indices[mid] < rid + 1) lo = mid + 1; else hi = mid; }
    const int end = lo;

    // trans = exp(-exclusive cumsum(s)) maintained multiplicatively.
    float trans = 1.f, cr = 0.f, cg = 0.f, cb = 0.f, acc = 0.f;
    for (int j = start; j < end; ++j) {
        const float4 v = ws[j];
        const float e = __expf(-v.x);
        const float w = trans * (1.0f - e);
        cr += w * v.y; cg += w * v.z; cb += w * v.w; acc += w;
        trans *= e;
    }
    const float bg = 1.0f - acc;                   // white background
    out[rid * 4 + 0] = fminf(fmaxf(cr + bg, 0.0f), 1.0f);
    out[rid * 4 + 1] = fminf(fmaxf(cg + bg, 0.0f), 1.0f);
    out[rid * 4 + 2] = fminf(fmaxf(cb + bg, 0.0f), 1.0f);
    out[rid * 4 + 3] = fminf(fmaxf(acc, 0.0f), 1.0f);
}

extern "C" void kernel_launch(void* const* d_in, const int* in_sizes, int n_in,
                              void* d_out, int out_size, void* d_ws, size_t ws_size,
                              hipStream_t stream) {
    (void)in_sizes; (void)n_in; (void)out_size; (void)ws_size;
    const float* rays_o        = (const float*)d_in[0];
    const float* rays_d        = (const float*)d_in[1];
    const float* t_starts      = (const float*)d_in[2];
    const float* t_ends        = (const float*)d_in[3];
    const int*   ray_indices   = (const int*)d_in[4];
    const float* table_density = (const float*)d_in[5];
    const float* table_color   = (const float*)d_in[6];
    const float* w1_density    = (const float*)d_in[7];
    const float* w2_density    = (const float*)d_in[8];
    const float* w1_color      = (const float*)d_in[9];
    const float* w2_color      = (const float*)d_in[10];
    float*  out = (float*)d_out;
    float4* ws  = (float4*)d_ws;   // 524288 * 16B = 8 MB scratch: {s, r, g, b}

    nerf_sample_kernel<<<MSAMP / 256, 256, 0, stream>>>(
        rays_o, rays_d, t_starts, t_ends, ray_indices,
        table_density, table_color,
        w1_density, w2_density, w1_color, w2_color, ws);

    nerf_render_kernel<<<NRAYS_ / 256, 256, 0, stream>>>(ray_indices, ws, out);
}
